// AttentionModel_6468220748046
// MI455X (gfx1250) — matrix operand order repaired and checked by reading, not compiled, not run on hardware
//
#include <hip/hip_runtime.h>
#include <hip/hip_bf16.h>

#define SEQ   179
#define SP    192          // SEQ padded to multiple of 32 for P@V K-dim
#define EMB   256
#define HEADS 8
#define HD    32
#define BATCH 512
#define NROWS (BATCH*SEQ)  // 91648 = 16 * 5728 (exact 16-row tiling)
#define MT    (NROWS/16)   // 5728 row tiles
#define EPSLN 1e-5f
#define KPE   (-0.03597789207803197f)   // -ln(10000)/256
#define PESC  ((float)EMB/(float)SEQ)

typedef __attribute__((ext_vector_type(16))) _Float16 v16h;
typedef __attribute__((ext_vector_type(8)))  _Float16 v8h;
typedef __attribute__((ext_vector_type(8)))  float    v8f;

static __device__ __forceinline__ v16h cat8(v8h lo, v8h hi) {
  return __builtin_shufflevector(lo, hi, 0,1,2,3,4,5,6,7,8,9,10,11,12,13,14,15);
}
// A-operand loader: row-major f16 row, contiguous k order (k = 16h + s, B matches) for 16x16x32
// lanes 0-15 take K {base..base+15}; lanes 16-31 shift by 16.
static __device__ __forceinline__ v16h load_a16(const _Float16* p) {
  return *(const v16h*)(p);
}

// ---------------------------------------------------------------------------
// 1) Conv1d(k=8,s=4) + BatchNorm + ReLU + positional encoding -> f16 Xh
// ---------------------------------------------------------------------------
__global__ __launch_bounds__(256)
void conv_embed_kernel(const float* __restrict__ x,
                       const float* __restrict__ cw, const float* __restrict__ cb,
                       const float* __restrict__ bng, const float* __restrict__ bnb,
                       const float* __restrict__ bnm, const float* __restrict__ bnv,
                       _Float16* __restrict__ Xh) {
  const int bs = blockIdx.x;          // b*SEQ + s
  const int bat = bs / SEQ, s = bs % SEQ;
  const int e  = threadIdx.x;         // 0..255 output channel
  const float* xp = x + (size_t)bat*720 + s*4;
  const float* wp = cw + e*8;
  float acc = cb[e];
#pragma unroll
  for (int k = 0; k < 8; ++k) acc += wp[k] * xp[k];
  const float inv = rsqrtf(bnv[e] + EPSLN);
  acc = (acc - bnm[e]) * (bng[e] * inv) + bnb[e];
  acc = fmaxf(acc, 0.0f);
  const int i2 = (e >> 1) * 2;
  const float dv  = __expf((float)i2 * KPE);
  const float ang = (float)s * dv * PESC;
  const float pe  = (e & 1) ? __cosf(ang) : __sinf(ang);
  Xh[(size_t)bs * EMB + e] = (_Float16)(acc + pe);
}

// ---------------------------------------------------------------------------
// helpers: f32->f16 weight convert, f16 zero-fill
// ---------------------------------------------------------------------------
__global__ __launch_bounds__(256)
void cvt_f16_kernel(const float* __restrict__ src, _Float16* __restrict__ dst, int n) {
  int i = blockIdx.x * 256 + threadIdx.x;
  if (i < n) dst[i] = (_Float16)src[i];
}
__global__ __launch_bounds__(256)
void zero_f16_kernel(_Float16* __restrict__ p, int n) {
  int i = blockIdx.x * 256 + threadIdx.x;
  if (i < n) p[i] = (_Float16)0.0f;
}

// ---------------------------------------------------------------------------
// 2) Fused QKV projection: Y = X @ W^T for W in {wq,wk,wv}
//    One wave computes one 16x16 output tile; 8 K-chunks of 32; 24 WMMAs.
//    Q,K stored [b,h,s,d]; V stored transposed [b,h,d,t] (t padded to SP).
// ---------------------------------------------------------------------------
__global__ __launch_bounds__(32)
void qkv_proj_kernel(const _Float16* __restrict__ Xh,
                     const _Float16* __restrict__ Wq,
                     const _Float16* __restrict__ Wk,
                     const _Float16* __restrict__ Wv,
                     _Float16* __restrict__ Qh,
                     _Float16* __restrict__ Kh,
                     _Float16* __restrict__ Vt) {
  const int mt = blockIdx.x;          // 0..MT-1
  const int nt = blockIdx.y;          // 0..15
  const int lane = threadIdx.x;
  const int l16 = lane & 15;
  const int lh  = (lane < 16) ? 0 : 1;
  const int rowA = mt * 16 + l16;
  const int colB = nt * 16 + l16;
  const _Float16* arow = Xh + (size_t)rowA * EMB;
  const _Float16* wqr  = Wq + (size_t)colB * EMB;
  const _Float16* wkr  = Wk + (size_t)colB * EMB;
  const _Float16* wvr  = Wv + (size_t)colB * EMB;

  v8f cq = {}, ck = {}, cv = {};
#pragma unroll
  for (int kc = 0; kc < 8; ++kc) {
    const int k0 = kc * 32;
    if (kc < 7) __builtin_prefetch(arow + k0 + 32, 0, 3);   // WGP-scope prefetch
    v16h a  = load_a16(arow + k0 + lh * 16);
    v16h bq = *(const v16h*)(wqr + k0 + lh * 16);
    v16h bk = *(const v16h*)(wkr + k0 + lh * 16);
    v16h bv = *(const v16h*)(wvr + k0 + lh * 16);
    cq = __builtin_amdgcn_wmma_f32_16x16x32_f16(false, a, false, bq, (short)0, cq, false, false);
    ck = __builtin_amdgcn_wmma_f32_16x16x32_f16(false, a, false, bk, (short)0, ck, false, false);
    cv = __builtin_amdgcn_wmma_f32_16x16x32_f16(false, a, false, bv, (short)0, cv, false, false);
  }

  const int h = nt >> 1;
  const int d = ((nt & 1) << 4) + l16;
#pragma unroll
  for (int r = 0; r < 8; ++r) {
    const int m = mt * 16 + r + lh * 8;
    const int bat = m / SEQ, s = m % SEQ;
    const size_t qi = (((size_t)bat * HEADS + h) * SEQ + s) * HD + d;
    Qh[qi] = (_Float16)cq[r];
    Kh[qi] = (_Float16)ck[r];
    const size_t vi = (((size_t)bat * HEADS + h) * HD + d) * SP + s;
    Vt[vi] = (_Float16)cv[r];
  }
}

// ---------------------------------------------------------------------------
// 3) Attention core: one wave per (batch, 16-row s-tile).
//    Per head: scores (12 WMMA, K=D=32) -> scale*sw -> softmax -> P@V (12 WMMA)
//    Softmax/LN split 2 lanes per row so all 32 lanes stay busy.
//    Layer 1: +PE, write f16 Xh (next layer input).
//    Final layer: lnA2 then ln2, write f32 rows for pooling.
// ---------------------------------------------------------------------------
__global__ __launch_bounds__(32)
void attn_kernel(const _Float16* __restrict__ Qh,
                 const _Float16* __restrict__ Kh,
                 const _Float16* __restrict__ Vt,
                 const float* __restrict__ g1, const float* __restrict__ b1,
                 const float* __restrict__ g2, const float* __restrict__ b2,
                 _Float16* __restrict__ XhOut,
                 float* __restrict__ AttF,
                 int final_block) {
  __shared__ __align__(16) float    Ps[16][SP];   // raw scores, f32
  __shared__ __align__(16) _Float16 Pm[16][SP];   // softmax probs, f16 (A source)
  __shared__ __align__(16) float    Ob[16][EMB];  // attention output row-block
  __shared__ float redA[2][16], redB[2][16];      // 2-lane reductions
  __shared__ float mu[16], rs[16];

  const int st  = blockIdx.x;          // 0..11 (SP/16)
  const int bat = blockIdx.y;
  const int lane = threadIdx.x;
  const int l16 = lane & 15;
  const int lh  = (lane < 16) ? 0 : 1;
  const int s0  = st * 16;
  const float scale = 1.0f / 16.0f;    // EMB^-0.5
  const float NEG = -1.0e30f;

  for (int h = 0; h < HEADS; ++h) {
    const _Float16* qbase = Qh + ((size_t)bat * HEADS + h) * SEQ * HD;
    const _Float16* kbase = Kh + ((size_t)bat * HEADS + h) * SEQ * HD;
    const _Float16* vbase = Vt + ((size_t)bat * HEADS + h) * HD * SP;

    int sA = s0 + l16; if (sA >= SEQ) sA = SEQ - 1;          // clamp pad rows
    v16h aq = load_a16(qbase + (size_t)sA * HD + lh * 16);

    // ---- scores: Q Kt, K-dim = D = 32 (single WMMA per 16x16 tile) ----
    for (int tt = 0; tt < SP / 16; ++tt) {
      const int t  = tt * 16 + l16;
      const int tl = (t < SEQ) ? t : (SEQ - 1);
      v16h bk = *(const v16h*)(kbase + (size_t)tl * HD + lh * 16);
      v8f c = {};
      c = __builtin_amdgcn_wmma_f32_16x16x32_f16(false, aq, false, bk, (short)0, c, false, false);
#pragma unroll
      for (int r = 0; r < 8; ++r) {
        const int m  = r + lh * 8;
        const int sg = s0 + m;
        float v;
        if (t >= SEQ) {
          v = NEG;
        } else {
          const float sw = fabsf((float)(sg - t)) * (1.0f / (float)SEQ);
          v = c[r] * scale * sw;
        }
        Ps[m][tt * 16 + l16] = v;
      }
    }
    __syncthreads();

    // ---- row softmax: 2 lanes per row (lane m: cols 0..95, lane m+16: 96..191)
    {
      const int m = l16;
      const int c0 = lh * (SP / 2), c1 = c0 + SP / 2;
      float mx = NEG;
      for (int t = c0; t < c1; ++t) mx = fmaxf(mx, Ps[m][t]);
      redA[lh][m] = mx;
      __syncthreads();
      mx = fmaxf(redA[0][m], redA[1][m]);
      float sum = 0.0f;
      for (int t = c0; t < c1; ++t) { float ev = __expf(Ps[m][t] - mx); Ps[m][t] = ev; sum += ev; }
      __syncthreads();                 // all reads of redA done before reuse
      redA[lh][m] = sum;
      __syncthreads();
      const float inv = 1.0f / (redA[0][m] + redA[1][m]);
      for (int t = c0; t < c1; ++t) Pm[m][t] = (_Float16)(Ps[m][t] * inv);
    }
    __syncthreads();

    // ---- out = P @ V : N-tiles over d (2), K-chunks over t (6) ----
#pragma unroll
    for (int nt2 = 0; nt2 < 2; ++nt2) {
      const int d = nt2 * 16 + l16;
      v8f c = {};
#pragma unroll
      for (int kc = 0; kc < SP / 32; ++kc) {
        const int k0 = kc * 32;
        v16h ap = load_a16(&Pm[l16][k0 + lh * 16]);
        v16h bv = *(const v16h*)(vbase + (size_t)d * SP + k0 + lh * 16);
        c = __builtin_amdgcn_wmma_f32_16x16x32_f16(false, ap, false, bv, (short)0, c, false, false);
      }
#pragma unroll
      for (int r = 0; r < 8; ++r) Ob[r + lh * 8][h * HD + d] = c[r];
    }
    __syncthreads();
  }

  // ---- LayerNorm (lnA) stats: 2 lanes per row, 128 elems each ----
  {
    const int m = l16;
    const int e0 = lh * (EMB / 2), e1 = e0 + EMB / 2;
    float su = 0.0f, sq = 0.0f;
    for (int e = e0; e < e1; ++e) { const float v = Ob[m][e]; su += v; sq += v * v; }
    redA[lh][m] = su; redB[lh][m] = sq;
    __syncthreads();
    const float mean = (redA[0][m] + redA[1][m]) * (1.0f / EMB);
    const float var  = (redB[0][m] + redB[1][m]) * (1.0f / EMB) - mean * mean;
    mu[m] = mean; rs[m] = rsqrtf(var + EPSLN);
  }
  __syncthreads();

  if (!final_block) {
    // layer-1 epilogue: y = LN(out)*g+b + PE  -> f16 next-layer input
    for (int e = lane; e < EMB; e += 32) {
      const float g = g1[e], bb = b1[e];
      const int i2 = (e >> 1) * 2;
      const float co = __expf((float)i2 * KPE) * PESC;
      for (int m = 0; m < 16; ++m) {
        const int s = s0 + m;
        if (s >= SEQ) break;
        const float y  = (Ob[m][e] - mu[m]) * rs[m] * g + bb;
        const float ang = (float)s * co;
        const float pe  = (e & 1) ? __cosf(ang) : __sinf(ang);
        XhOut[((size_t)bat * SEQ + s) * EMB + e] = (_Float16)(y + pe);
      }
    }
  } else {
    // final epilogue: lnA2 in-place, then ln2, write f32 rows
    for (int e = lane; e < EMB; e += 32) {
      const float g = g1[e], bb = b1[e];
      for (int m = 0; m < 16; ++m)
        Ob[m][e] = (Ob[m][e] - mu[m]) * rs[m] * g + bb;
    }
    __syncthreads();
    {
      const int m = l16;
      const int e0 = lh * (EMB / 2), e1 = e0 + EMB / 2;
      float su = 0.0f, sq = 0.0f;
      for (int e = e0; e < e1; ++e) { const float v = Ob[m][e]; su += v; sq += v * v; }
      redA[lh][m] = su; redB[lh][m] = sq;
      __syncthreads();
      const float mean = (redA[0][m] + redA[1][m]) * (1.0f / EMB);
      const float var  = (redB[0][m] + redB[1][m]) * (1.0f / EMB) - mean * mean;
      mu[m] = mean; rs[m] = rsqrtf(var + EPSLN);
    }
    __syncthreads();
    for (int e = lane; e < EMB; e += 32) {
      const float g = g2[e], bb = b2[e];
      for (int m = 0; m < 16; ++m) {
        const int s = s0 + m;
        if (s >= SEQ) break;
        AttF[((size_t)bat * SEQ + s) * EMB + e] = (Ob[m][e] - mu[m]) * rs[m] * g + bb;
      }
    }
  }
}

// ---------------------------------------------------------------------------
// 4) Global average pool over SEQ + 10-way linear head
// ---------------------------------------------------------------------------
__global__ __launch_bounds__(256)
void pool_fc_kernel(const float* __restrict__ AttF,
                    const float* __restrict__ ow, const float* __restrict__ ob,
                    float* __restrict__ out) {
  __shared__ float pooled[EMB];
  const int bat = blockIdx.x, e = threadIdx.x;
  float s = 0.0f;
  for (int t = 0; t < SEQ; ++t) s += AttF[((size_t)bat * SEQ + t) * EMB + e];
  pooled[e] = s * (1.0f / (float)SEQ);
  __syncthreads();
  if (e < 10) {
    float acc = ob[e];
    for (int k = 0; k < EMB; ++k) acc += pooled[k] * ow[e * EMB + k];
    out[bat * 10 + e] = acc;
  }
}

// ---------------------------------------------------------------------------
extern "C" void kernel_launch(void* const* d_in, const int* in_sizes, int n_in,
                              void* d_out, int out_size, void* d_ws, size_t ws_size,
                              hipStream_t stream) {
  (void)in_sizes; (void)n_in; (void)out_size; (void)ws_size;
  const float* x      = (const float*)d_in[0];
  const float* conv_w = (const float*)d_in[1];
  const float* conv_b = (const float*)d_in[2];
  const float* bn_g   = (const float*)d_in[3];
  const float* bn_b   = (const float*)d_in[4];
  const float* bn_m   = (const float*)d_in[5];
  const float* bn_v   = (const float*)d_in[6];
  const float* wq1    = (const float*)d_in[7];
  const float* wk1    = (const float*)d_in[8];
  const float* wv1    = (const float*)d_in[9];
  const float* lnA1_g = (const float*)d_in[10];
  const float* lnA1_b = (const float*)d_in[11];
  const float* wq2    = (const float*)d_in[12];
  const float* wk2    = (const float*)d_in[13];
  const float* wv2    = (const float*)d_in[14];
  const float* lnA2_g = (const float*)d_in[15];
  const float* lnA2_b = (const float*)d_in[16];
  const float* ln2_g  = (const float*)d_in[17];
  const float* ln2_b  = (const float*)d_in[18];
  const float* out_w  = (const float*)d_in[19];
  const float* out_b  = (const float*)d_in[20];
  float* out = (float*)d_out;

  // workspace carve-up (256B aligned)
  char* ws = (char*)d_ws;
  size_t off = 0;
  auto take = [&](size_t bytes) -> char* {
    char* p = ws + off;
    off += (bytes + 255) & ~(size_t)255;
    return p;
  };
  _Float16* Xh  = (_Float16*)take((size_t)NROWS * EMB * 2);
  _Float16* Qh  = (_Float16*)take((size_t)NROWS * EMB * 2);
  _Float16* Kh  = (_Float16*)take((size_t)NROWS * EMB * 2);
  _Float16* Vt  = (_Float16*)take((size_t)BATCH * HEADS * HD * SP * 2);
  _Float16* Wh  = (_Float16*)take((size_t)6 * EMB * EMB * 2);
  float*    AttF= (float*)   take((size_t)NROWS * EMB * 4);

  const int WN = EMB * EMB;   // 65536

  // 1) conv + BN + ReLU + PE
  conv_embed_kernel<<<NROWS, 256, 0, stream>>>(x, conv_w, conv_b, bn_g, bn_b, bn_m, bn_v, Xh);

  // 2) weight converts + V pad zero
  cvt_f16_kernel<<<WN / 256, 256, 0, stream>>>(wq1, Wh + 0 * WN, WN);
  cvt_f16_kernel<<<WN / 256, 256, 0, stream>>>(wk1, Wh + 1 * WN, WN);
  cvt_f16_kernel<<<WN / 256, 256, 0, stream>>>(wv1, Wh + 2 * WN, WN);
  cvt_f16_kernel<<<WN / 256, 256, 0, stream>>>(wq2, Wh + 3 * WN, WN);
  cvt_f16_kernel<<<WN / 256, 256, 0, stream>>>(wk2, Wh + 4 * WN, WN);
  cvt_f16_kernel<<<WN / 256, 256, 0, stream>>>(wv2, Wh + 5 * WN, WN);
  const int vtN = BATCH * HEADS * HD * SP;
  zero_f16_kernel<<<(vtN + 255) / 256, 256, 0, stream>>>(Vt, vtN);

  // 3) layer 1
  qkv_proj_kernel<<<dim3(MT, 16), 32, 0, stream>>>(Xh, Wh + 0 * WN, Wh + 1 * WN, Wh + 2 * WN, Qh, Kh, Vt);
  attn_kernel<<<dim3(SP / 16, BATCH), 32, 0, stream>>>(Qh, Kh, Vt, lnA1_g, lnA1_b, ln2_g, ln2_b, Xh, AttF, 0);

  // 4) layer 2 (+ final ln2, writes f32 rows)
  qkv_proj_kernel<<<dim3(MT, 16), 32, 0, stream>>>(Xh, Wh + 3 * WN, Wh + 4 * WN, Wh + 5 * WN, Qh, Kh, Vt);
  attn_kernel<<<dim3(SP / 16, BATCH), 32, 0, stream>>>(Qh, Kh, Vt, lnA2_g, lnA2_b, ln2_g, ln2_b, Xh, AttF, 1);

  // 5) pooling + classifier head
  pool_fc_kernel<<<BATCH, 256, 0, stream>>>(AttF, out_w, out_b, out);
}
